// CoGAT_75376676045427
// MI455X (gfx1250) — matrix operand order repaired and checked by reading, not compiled, not run on hardware
//
#include <hip/hip_runtime.h>
#include <cmath>

// ---------------- problem constants ----------------
#define NNODE 3000
#define NP    3072          // padded node dim (divisible by 64 and 32)
#define EDGES 96000
#define INDIM 1280
#define DDIM  128
#define KCHEB 24            // Chebyshev degree for exp(-2L) and exp(-4 exp(-2L))

typedef __attribute__((ext_vector_type(16))) __bf16 v16bf;
typedef __attribute__((ext_vector_type(8)))  float  v8f;

// Async direct-to-LDS staging (gfx1250): guarded so the proven register
// pipeline is kept verbatim if the toolchain lacks the builtins.
#if __has_builtin(__builtin_amdgcn_global_load_async_to_lds_b128) && \
    __has_builtin(__builtin_amdgcn_s_wait_asynccnt)
#define USE_ASYNC 1
// exact parameter type per clang diagnostic: int __attribute__((vector_size(16)))
typedef int v4i_vs __attribute__((__vector_size__(4 * sizeof(int))));
typedef __attribute__((address_space(1))) v4i_vs g1v4i;
typedef __attribute__((address_space(3))) v4i_vs l3v4i;
// flat LDS address low 32 bits == LDS byte offset (CDNA5 aperture rule, ISA 10.2);
// AS3 pointers are 32-bit so the inttoptr truncation lands on the LDS offset.
#define GASYNC_B128(g, l)                                              \
  __builtin_amdgcn_global_load_async_to_lds_b128(                      \
      (g1v4i*)(uintptr_t)(g), (l3v4i*)(uintptr_t)(l), 0, 0)
#else
#define USE_ASYNC 0
#endif

__device__ __forceinline__ unsigned short f2b(float f) {
  unsigned int u = __float_as_uint(f);
  u += 0x7FFFu + ((u >> 16) & 1u);          // round-to-nearest-even
  return (unsigned short)(u >> 16);
}
__device__ __forceinline__ unsigned pack2(float a, float b) {
  return (unsigned)f2b(a) | ((unsigned)f2b(b) << 16);
}
__device__ __forceinline__ unsigned enc_f(float f) {   // order-preserving f32->u32
  unsigned u = __float_as_uint(f);
  return (u & 0x80000000u) ? ~u : (u | 0x80000000u);
}
__device__ __forceinline__ float dec_f(unsigned e) {
  unsigned u = (e & 0x80000000u) ? (e & 0x7FFFFFFFu) : ~e;
  return __uint_as_float(u);
}

// A operand layout (u32 index): K-blocked rows of 16 u32 (= 32 bf16)
__device__ __forceinline__ size_t ablk(int m, int k, int M) {
  return ((size_t)(k >> 5) * M + m) * 16 + ((k >> 1) & 15);
}
// B operand layout (u32 index): fragment-linear [k/32][nt][lane][p]
__device__ __forceinline__ size_t bswz(int k, int n) {
  return ((size_t)(k >> 5) * 8 + (n >> 4)) * 256 +
         ((((k >> 4) & 1) * 16 + (n & 15)) * 8) + ((k >> 1) & 7);
}

// ---------------- WMMA GEMM: C[M,128] (+)= A[M,K] * B[K,128], bf16 operands ----------------
// 128 threads (4 waves); tile 64x128; BK=32; double-buffered LDS.
// flags: 1 = accumulate into C, 2 = relu, 4 = atomic split-K reduction (no bias/relu/Cbsw)
#define GBM 64
#define GBK 32

__launch_bounds__(128)
__global__ void gemm_bf16(const unsigned int* __restrict__ A,   // A-blocked
                          const unsigned int* __restrict__ B,   // B fragment-linear
                          float* __restrict__ C,
                          unsigned int* __restrict__ Cbsw,      // optional bf16 out, B-layout
                          const float* __restrict__ bias,
                          int M, int K, int flags)
{
  __shared__ uint4 As[2][256];   // 2 x 4KB
  __shared__ uint4 Bs[2][512];   // 2 x 8KB

  const int t = threadIdx.x, wave = t >> 5, lane = t & 31;
  const int half = lane >> 4, l15 = lane & 15;
  const int m0 = blockIdx.x * GBM;

  const int kchunk = K / gridDim.z;
  const int kbeg = blockIdx.z * kchunk, kend = kbeg + kchunk;

  const uint4* Ag = (const uint4*)A;
  const uint4* Bg = (const uint4*)B;

  v8f acc[8] = {};

#if USE_ASYNC
  // ---- async direct-to-LDS pipeline ----
  {
    const uint4* at = Ag + ((size_t)(kbeg >> 5) * M + m0) * 4;
    const uint4* bt = Bg + (size_t)(kbeg >> 5) * 512;
    GASYNC_B128(at + t,       &As[0][t]);
    GASYNC_B128(at + t + 128, &As[0][t + 128]);
    GASYNC_B128(bt + t,       &Bs[0][t]);
    GASYNC_B128(bt + t + 128, &Bs[0][t + 128]);
    GASYNC_B128(bt + t + 256, &Bs[0][t + 256]);
    GASYNC_B128(bt + t + 384, &Bs[0][t + 384]);
  }
  int buf = 0;
  for (int k0 = kbeg; k0 < kend; k0 += GBK) {
    __builtin_amdgcn_s_wait_asynccnt(0);   // current buffer landed in LDS
    __syncthreads();                       // publish to all waves
    const int k1 = k0 + GBK;
    if (k1 < kend) {                       // copy next tile while computing
      const uint4* at = Ag + ((size_t)(k1 >> 5) * M + m0) * 4;
      const uint4* bt = Bg + (size_t)(k1 >> 5) * 512;
      GASYNC_B128(at + t,       &As[buf ^ 1][t]);
      GASYNC_B128(at + t + 128, &As[buf ^ 1][t + 128]);
      GASYNC_B128(bt + t,       &Bs[buf ^ 1][t]);
      GASYNC_B128(bt + t + 128, &Bs[buf ^ 1][t + 128]);
      GASYNC_B128(bt + t + 256, &Bs[buf ^ 1][t + 256]);
      GASYNC_B128(bt + t + 384, &Bs[buf ^ 1][t + 384]);
    }
    if (k0 + 2 * GBK < kend) {             // L2 prefetch of tile k+2
      __builtin_prefetch((const char*)(Ag + ((size_t)((k0 + 2 * GBK) >> 5) * M + m0) * 4) + (t << 4), 0, 1);
      __builtin_prefetch((const char*)(Bg + (size_t)((k0 + 2 * GBK) >> 5) * 512) + (t << 5), 0, 1);
    }
    const uint4* As4 = As[buf];
    const uint4* Bf  = Bs[buf];
    union { v16bf v; uint4 q[2]; } af;
    af.q[0] = As4[(wave * 16 + l15) * 4 + half * 2];
    af.q[1] = As4[(wave * 16 + l15) * 4 + half * 2 + 1];
#pragma unroll
    for (int nt = 0; nt < 8; nt++) {
      union { v16bf v; uint4 q[2]; } bfu;
      bfu.q[0] = Bf[nt * 64 + lane * 2];
      bfu.q[1] = Bf[nt * 64 + lane * 2 + 1];
      acc[nt] = __builtin_amdgcn_wmma_f32_16x16x32_bf16(
          false, af.v, false, bfu.v, (short)0, acc[nt], false, false);
    }
    buf ^= 1;
  }
#else
  // ---- register-staged pipeline (fallback) ----
  uint4 ra0, ra1, rb0, rb1, rb2, rb3;
  {
    const uint4* at = Ag + ((size_t)(kbeg >> 5) * M + m0) * 4;
    const uint4* bt = Bg + (size_t)(kbeg >> 5) * 512;
    ra0 = at[t]; ra1 = at[t + 128];
    rb0 = bt[t]; rb1 = bt[t + 128]; rb2 = bt[t + 256]; rb3 = bt[t + 384];
  }
  int buf = 0;
  for (int k0 = kbeg; k0 < kend; k0 += GBK) {
    As[buf][t] = ra0; As[buf][t + 128] = ra1;
    Bs[buf][t] = rb0; Bs[buf][t + 128] = rb1;
    Bs[buf][t + 256] = rb2; Bs[buf][t + 384] = rb3;
    __syncthreads();
    const int k1 = k0 + GBK;
    if (k1 < kend) {
      const uint4* at = Ag + ((size_t)(k1 >> 5) * M + m0) * 4;
      const uint4* bt = Bg + (size_t)(k1 >> 5) * 512;
      ra0 = at[t]; ra1 = at[t + 128];
      rb0 = bt[t]; rb1 = bt[t + 128]; rb2 = bt[t + 256]; rb3 = bt[t + 384];
    }
    if (k0 + 2 * GBK < kend) {
      __builtin_prefetch((const char*)(Ag + ((size_t)((k0 + 2 * GBK) >> 5) * M + m0) * 4) + (t << 4), 0, 1);
      __builtin_prefetch((const char*)(Bg + (size_t)((k0 + 2 * GBK) >> 5) * 512) + (t << 5), 0, 1);
    }
    const uint4* As4 = As[buf];
    const uint4* Bf  = Bs[buf];
    union { v16bf v; uint4 q[2]; } af;
    af.q[0] = As4[(wave * 16 + l15) * 4 + half * 2];
    af.q[1] = As4[(wave * 16 + l15) * 4 + half * 2 + 1];
#pragma unroll
    for (int nt = 0; nt < 8; nt++) {
      union { v16bf v; uint4 q[2]; } bfu;
      bfu.q[0] = Bf[nt * 64 + lane * 2];
      bfu.q[1] = Bf[nt * 64 + lane * 2 + 1];
      acc[nt] = __builtin_amdgcn_wmma_f32_16x16x32_bf16(
          false, af.v, false, bfu.v, (short)0, acc[nt], false, false);
    }
    buf ^= 1;
  }
#endif

  // ---- epilogue ----
#pragma unroll
  for (int nt = 0; nt < 8; nt++) {
    const int col = nt * 16 + l15;
    if (flags & 4) {
#pragma unroll
      for (int r = 0; r < 8; r++) {
        int row = m0 + wave * 16 + 8 * half + r;
        atomicAdd(&C[(size_t)row * DDIM + col], acc[nt][r]);
      }
    } else {
      float fv[8];
      const float bb = bias ? bias[col] : 0.f;
#pragma unroll
      for (int r = 0; r < 8; r++) {
        int row = m0 + wave * 16 + 8 * half + r;
        size_t idx = (size_t)row * DDIM + col;
        float v = acc[nt][r];
        if (flags & 1) v += C[idx];
        v += bb;
        if (flags & 2) v = fmaxf(v, 0.f);
        C[idx] = v;
        fv[r] = v;
      }
      if (Cbsw) {
#pragma unroll
        for (int r = 0; r < 8; r += 2) {
          int row = m0 + wave * 16 + 8 * half + r;   // even
          Cbsw[bswz(row, col)] = pack2(fv[r], fv[r + 1]);
        }
      }
    }
  }
}

// ---------------- setup / layout-conversion kernels ----------------
__global__ void rowsum_rsqrt(const float* __restrict__ adj, float* __restrict__ dinv) {
  __shared__ float red[256];
  int row = blockIdx.x;
  float s = 0.f;
  for (int j = threadIdx.x; j < NNODE; j += 256) s += adj[(size_t)row * NNODE + j];
  red[threadIdx.x] = s; __syncthreads();
  for (int o = 128; o; o >>= 1) {
    if (threadIdx.x < o) red[threadIdx.x] += red[threadIdx.x + o];
    __syncthreads();
  }
  if (!threadIdx.x) dinv[row] = rsqrtf(red[0]);
}

__global__ void build_ahat(const float* __restrict__ adj, const float* __restrict__ dinv,
                           unsigned int* __restrict__ Ab) {
  size_t idx = (size_t)blockIdx.x * 256 + threadIdx.x;
  if (idx >= (size_t)NP * NP / 2) return;
  int i = (int)(idx / (NP / 2)), kp = (int)(idx % (NP / 2));
  int k = 2 * kp;
  float v0 = 0.f, v1 = 0.f;
  if (i < NNODE) {
    float di = dinv[i];
    if (k < NNODE)     v0 = di * adj[(size_t)i * NNODE + k]     * dinv[k];
    if (k + 1 < NNODE) v1 = di * adj[(size_t)i * NNODE + k + 1] * dinv[k + 1];
  }
  Ab[ablk(i, k, NP)] = pack2(v0, v1);
}

__global__ void pad_feat(const float* __restrict__ f, unsigned int* __restrict__ fb) {
  size_t idx = (size_t)blockIdx.x * 256 + threadIdx.x;
  if (idx >= (size_t)NP * INDIM / 2) return;
  int i = (int)(idx / (INDIM / 2)), kp = (int)(idx % (INDIM / 2));
  int k = 2 * kp;
  float v0 = 0.f, v1 = 0.f;
  if (i < NNODE) {
    v0 = f[(size_t)i * INDIM + k];
    v1 = f[(size_t)i * INDIM + k + 1];
  }
  fb[ablk(i, k, NP)] = pack2(v0, v1);
}

__global__ void wt_bswz(const float* __restrict__ W, unsigned int* __restrict__ Wt) {
  int idx = blockIdx.x * 256 + threadIdx.x;
  if (idx >= INDIM * DDIM / 2) return;
  int n = idx / (INDIM / 2), kp = idx % (INDIM / 2);
  int k = 2 * kp;
  Wt[bswz(k, n)] = pack2(W[n * INDIM + k], W[n * INDIM + k + 1]);
}

__global__ void cvt_bswz128(const float* __restrict__ s, unsigned int* __restrict__ d) {
  int idx = blockIdx.x * 256 + threadIdx.x;
  if (idx >= DDIM * DDIM / 2) return;
  int kp = idx >> 7, n = idx & 127;
  int k = 2 * kp;
  d[bswz(k, n)] = pack2(s[k * DDIM + n], s[(k + 1) * DDIM + n]);
}

__global__ void build_ml(const float* __restrict__ wl, unsigned int* __restrict__ Ml, float beta) {
  int idx = blockIdx.x * 256 + threadIdx.x;
  if (idx >= DDIM * DDIM / 2) return;
  int kp = idx >> 7, n = idx & 127;
  int k = 2 * kp;
  float v0 = beta * wl[k * DDIM + n]       + ((k == n)     ? (1.f - beta) : 0.f);
  float v1 = beta * wl[(k + 1) * DDIM + n] + ((k + 1 == n) ? (1.f - beta) : 0.f);
  Ml[bswz(k, n)] = pack2(v0, v1);
}

__global__ void cvt_ablk(const float* __restrict__ s, unsigned int* __restrict__ d) {
  int idx = blockIdx.x * 256 + threadIdx.x;
  if (idx >= (int)((size_t)NP * DDIM / 2)) return;
  int m = idx >> 6, kp = idx & 63;
  int k = 2 * kp;
  d[ablk(m, k, NP)] = pack2(s[(size_t)m * DDIM + k], s[(size_t)m * DDIM + k + 1]);
}

__global__ void mix_ablk(const float* __restrict__ hout, const float* __restrict__ h0,
                         unsigned int* __restrict__ d) {
  int idx = blockIdx.x * 256 + threadIdx.x;
  if (idx >= (int)((size_t)NP * DDIM / 2)) return;
  int m = idx >> 6, kp = idx & 63;
  int k = 2 * kp;
  size_t b = (size_t)m * DDIM + k;
  float v0 = 0.3f * hout[b]     + 0.7f * h0[b];
  float v1 = 0.3f * hout[b + 1] + 0.7f * h0[b + 1];
  d[ablk(m, k, NP)] = pack2(v0, v1);
}

__global__ void compute_wprod(const float* __restrict__ we, float* __restrict__ wprod) {
  if (threadIdx.x == 0 && blockIdx.x == 0) {
    float pr = 1.f;
    for (int l = 0; l < 7; l++) { pr *= we[l]; wprod[l] = pr; }
  }
}
__global__ void zero_f32(float* __restrict__ p, long n) {
  long i = (long)blockIdx.x * 256 + threadIdx.x;
  if (i < n) p[i] = 0.f;
}
__global__ void fill_u32(unsigned* __restrict__ p, unsigned v, long n) {
  long i = (long)blockIdx.x * 256 + threadIdx.x;
  if (i < n) p[i] = v;
}

// ---------------- Chebyshev kernels ----------------
__global__ void cheb_init(const float* __restrict__ h, float* __restrict__ Y1,
                          float* __restrict__ Y2, float c1, float c2, long n) {
  long i = (long)blockIdx.x * 256 + threadIdx.x;
  if (i < n) { float v = h[i]; Y1[i] = c1 * v; Y2[i] = c2 * v; }
}

__global__ void cheb_step(const float* __restrict__ S, const float* __restrict__ Tprev,
                          float* __restrict__ Tnew, unsigned int* __restrict__ Tnewb,
                          float* __restrict__ Y1, float* __restrict__ Y2,
                          float sA, float sP, float c1, float c2) {
  int idx = blockIdx.x * 256 + threadIdx.x;
  if (idx >= (int)((size_t)NP * DDIM / 2)) return;
  int kp = idx >> 7, n = idx & 127;
  int k = 2 * kp;
  size_t i0 = (size_t)k * DDIM + n, i1 = i0 + DDIM;
  float t0 = sA * S[i0] + sP * Tprev[i0];
  float t1 = sA * S[i1] + sP * Tprev[i1];
  Tnew[i0] = t0; Tnew[i1] = t1;
  Y1[i0] += c1 * t0; Y1[i1] += c1 * t1;
  Y2[i0] += c2 * t0; Y2[i1] += c2 * t1;
  Tnewb[bswz(k, n)] = pack2(t0, t1);
}

// ---------------- attention kernels ----------------
__global__ void node_pq(const float* __restrict__ z, const float* __restrict__ attnl,
                        float* __restrict__ p, float* __restrict__ q) {
  int node = blockIdx.x * 8 + (threadIdx.x >> 5);
  int lane = threadIdx.x & 31;
  if (node >= NNODE) return;
  float sp = 0.f, sq = 0.f;
#pragma unroll
  for (int j = 0; j < 4; j++) {
    int d = lane + 32 * j;
    float zv = z[(size_t)node * DDIM + d];
    sp += zv * attnl[d];
    sq += zv * attnl[DDIM + d];
  }
  for (int o = 16; o; o >>= 1) { sp += __shfl_xor(sp, o, 32); sq += __shfl_xor(sq, o, 32); }
  if (!lane) { p[node] = sp; q[node] = sq; }
}

__global__ void edge_score(const int* __restrict__ src, const int* __restrict__ dst,
                           const float* __restrict__ e, const float* __restrict__ wprod, int l,
                           const float* __restrict__ p, const float* __restrict__ q,
                           const float* __restrict__ attnl,
                           float* __restrict__ score, unsigned* __restrict__ menc) {
  int ei = blockIdx.x * 256 + threadIdx.x;
  if (ei >= EDGES) return;
  float exv = e[ei] * wprod[l];
  float a = p[src[ei]] + q[dst[ei]] + exv * attnl[2 * DDIM];
  float s = a > 0.f ? a : 0.01f * a;
  score[ei] = s;
  atomicMax(&menc[dst[ei]], enc_f(s));
}

__global__ void edge_exp(const int* __restrict__ dst, const float* __restrict__ score,
                         const unsigned* __restrict__ menc,
                         float* __restrict__ ew, float* __restrict__ denom) {
  int ei = blockIdx.x * 256 + threadIdx.x;
  if (ei >= EDGES) return;
  float m = dec_f(menc[dst[ei]]);
  float v = __expf(score[ei] - m);
  ew[ei] = v;
  atomicAdd(&denom[dst[ei]], v);
}

__global__ void edge_attn(const int* __restrict__ dst, const float* __restrict__ ew,
                          const float* __restrict__ denom, float* __restrict__ attn_e) {
  int ei = blockIdx.x * 256 + threadIdx.x;
  if (ei >= EDGES) return;
  attn_e[ei] = ew[ei] / denom[dst[ei]];
}

__global__ void edge_scatter(const int* __restrict__ src, const int* __restrict__ dst,
                             const float* __restrict__ e, const float* __restrict__ wprod, int l,
                             const float* __restrict__ attn_e, const float* __restrict__ z,
                             const float* __restrict__ wezl, float* __restrict__ hout) {
  long idx = (long)blockIdx.x * 256 + threadIdx.x;
  if (idx >= (long)EDGES * DDIM) return;
  int ei = (int)(idx >> 7), d = (int)(idx & 127);
  float exv = e[ei] * wprod[l];
  float v = attn_e[ei] * (z[(size_t)src[ei] * DDIM + d] + exv * wezl[d]);
  atomicAdd(&hout[(size_t)dst[ei] * DDIM + d], v);
}

__global__ void copy_out(const float* __restrict__ hf, float* __restrict__ out) {
  long i = (long)blockIdx.x * 256 + threadIdx.x;
  if (i < (long)NNODE * DDIM) out[i] = hf[i];
}

// ---------------- host driver ----------------
extern "C" void kernel_launch(void* const* d_in, const int* in_sizes, int n_in,
                              void* d_out, int out_size, void* d_ws, size_t ws_size,
                              hipStream_t stream) {
  const float* features = (const float*)d_in[0];
  const float* e_arr    = (const float*)d_in[1];
  const float* adj      = (const float*)d_in[2];
  const int*   src      = (const int*)d_in[3];
  const int*   dst      = (const int*)d_in[4];
  const float* W        = (const float*)d_in[5];
  const float* bvec     = (const float*)d_in[6];
  const float* wl       = (const float*)d_in[7];
  const float* gw1      = (const float*)d_in[8];
  const float* gw2      = (const float*)d_in[9];
  const float* gw3      = (const float*)d_in[10];
  const float* gb       = (const float*)d_in[11];
  const float* attn     = (const float*)d_in[12];
  const float* we       = (const float*)d_in[13];
  const float* wez      = (const float*)d_in[14];

  // -------- workspace carve (~50 MB) --------
  char* wsp = (char*)d_ws;
  auto carve = [&](size_t bytes) -> char* {
    char* p = wsp; wsp += (bytes + 255) & ~(size_t)255; return p;
  };
  const long ND = (long)NP * DDIM;
  unsigned* Ahatb  = (unsigned*)carve((size_t)NP * NP * 2);
  unsigned* featb  = (unsigned*)carve((size_t)NP * INDIM * 2);
  unsigned* Wtbsw  = (unsigned*)carve((size_t)INDIM * DDIM * 2);
  unsigned* gw1bsw = (unsigned*)carve((size_t)7 * DDIM * DDIM * 2);
  unsigned* gw2bsw = (unsigned*)carve((size_t)7 * DDIM * DDIM * 2);
  unsigned* gw3bsw = (unsigned*)carve((size_t)7 * DDIM * DDIM * 2);
  unsigned* Mlbsw  = (unsigned*)carve((size_t)DDIM * DDIM * 2);
  float* h0f = (float*)carve(ND * 4);
  unsigned* h0bsw  = (unsigned*)carve(ND * 2);
  unsigned* h0ablk = (unsigned*)carve(ND * 2);
  float* hf  = (float*)carve(ND * 4);
  unsigned* hbsw  = (unsigned*)carve(ND * 2);
  unsigned* hablk = (unsigned*)carve(ND * 2);
  unsigned* mixb  = (unsigned*)carve(ND * 2);
  float* Sf  = (float*)carve(ND * 4);
  float* T1f = (float*)carve(ND * 4); unsigned* T1bsw = (unsigned*)carve(ND * 2);
  float* T2f = (float*)carve(ND * 4); unsigned* T2bsw = (unsigned*)carve(ND * 2);
  float* Y1  = (float*)carve(ND * 4); unsigned* Y1ablk = (unsigned*)carve(ND * 2);
  float* Y2  = (float*)carve(ND * 4); unsigned* Y2ablk = (unsigned*)carve(ND * 2);
  float* zf   = (float*)carve(ND * 4);
  float* hout = (float*)carve(ND * 4);
  float* dinv = (float*)carve((size_t)NP * 4);
  float* pbuf = (float*)carve((size_t)NNODE * 4);
  float* qbuf = (float*)carve((size_t)NNODE * 4);
  float* score  = (float*)carve((size_t)EDGES * 4);
  float* ewbuf  = (float*)carve((size_t)EDGES * 4);
  float* attn_e = (float*)carve((size_t)EDGES * 4);
  unsigned* menc = (unsigned*)carve((size_t)NNODE * 4);
  float* denom = (float*)carve((size_t)NNODE * 4);
  float* wprod = (float*)carve(8 * 4);

  // -------- Chebyshev coefficients: f1(x)=exp(-2(x+1)), f2(x)=exp(-4 exp(-2(x+1))) --------
  float c1[KCHEB], c2[KCHEB];
  {
    const int M = 256;
    for (int j = 0; j < KCHEB; j++) {
      double s1 = 0.0, s2 = 0.0;
      for (int m = 0; m < M; m++) {
        double th = M_PI * (m + 0.5) / M;
        double x  = cos(th);
        s1 += exp(-2.0 * (x + 1.0)) * cos(j * th);
        s2 += exp(-4.0 * exp(-2.0 * (x + 1.0))) * cos(j * th);
      }
      double sc = (j == 0 ? 1.0 : 2.0) / M;
      c1[j] = (float)(sc * s1); c2[j] = (float)(sc * s2);
    }
  }

  auto gemm = [&](const unsigned* A, const unsigned* B, float* C, unsigned* Cbsw,
                  const float* bias, int M, int K, int flags, int splitK) {
    dim3 g(M / GBM, 1, splitK);
    gemm_bf16<<<g, 128, 0, stream>>>(A, B, C, Cbsw, bias, M, K, flags);
  };
  const int EB  = EDGES / 256;             // 375
  const int NDB = (int)(ND / 256);         // 1536
  const int HDB = (int)(ND / 512);         // 768 (pair kernels)

  // -------- setup --------
  rowsum_rsqrt<<<NNODE, 256, 0, stream>>>(adj, dinv);
  build_ahat<<<(int)(((size_t)NP * NP / 2) / 256), 256, 0, stream>>>(adj, dinv, Ahatb);
  pad_feat<<<(int)(((size_t)NP * INDIM / 2) / 256), 256, 0, stream>>>(features, featb);
  wt_bswz<<<(INDIM * DDIM / 2) / 256, 256, 0, stream>>>(W, Wtbsw);
  for (int l = 0; l < 7; l++) {
    cvt_bswz128<<<32, 256, 0, stream>>>(gw1 + (size_t)l * DDIM * DDIM, gw1bsw + (size_t)l * DDIM * DDIM / 2);
    cvt_bswz128<<<32, 256, 0, stream>>>(gw2 + (size_t)l * DDIM * DDIM, gw2bsw + (size_t)l * DDIM * DDIM / 2);
    cvt_bswz128<<<32, 256, 0, stream>>>(gw3 + (size_t)l * DDIM * DDIM, gw3bsw + (size_t)l * DDIM * DDIM / 2);
  }
  compute_wprod<<<1, 32, 0, stream>>>(we, wprod);

  // h0 = features @ W.T + b
  gemm(featb, Wtbsw, h0f, h0bsw, bvec, NP, INDIM, 0, 1);
  cvt_ablk<<<HDB, 256, 0, stream>>>(h0f, h0ablk);

  // -------- layers --------
  for (int l = 0; l < 7; l++) {
    float beta = (float)log(1.5 / (l + 1) + 1.0);
    build_ml<<<32, 256, 0, stream>>>(wl + (size_t)l * DDIM * DDIM, Mlbsw, beta);

    const float*    h_f    = (l == 0) ? h0f : hf;
    const unsigned* h_bsw  = (l == 0) ? h0bsw : hbsw;
    const unsigned* h_ablk = (l == 0) ? h0ablk : hablk;

    // Chebyshev: T0 = h ; T1 = -A h ; T_{k+1} = -2 A T_k - T_{k-1}
    cheb_init<<<NDB, 256, 0, stream>>>(h_f, Y1, Y2, c1[0], c2[0], ND);
    const float*    prevf = h_f;
    const float*    curf  = h_f;
    const unsigned* curb  = h_bsw;
    for (int k = 1; k < KCHEB; k++) {
      zero_f32<<<NDB, 256, 0, stream>>>(Sf, ND);
      gemm(Ahatb, curb, Sf, nullptr, nullptr, NP, NP, 4, 8);   // split-K=8, atomic
      float* tf = (k & 1) ? T1f : T2f;
      unsigned* tb = (k & 1) ? T1bsw : T2bsw;
      float sA = (k == 1) ? -1.f : -2.f;
      float sP = (k == 1) ?  0.f : -1.f;
      cheb_step<<<HDB, 256, 0, stream>>>(Sf, prevf, tf, tb, Y1, Y2, sA, sP, c1[k], c2[k]);
      prevf = curf; curf = tf; curb = tb;
    }
    cvt_ablk<<<HDB, 256, 0, stream>>>(Y1, Y1ablk);
    cvt_ablk<<<HDB, 256, 0, stream>>>(Y2, Y2ablk);

    // z = h@gw1 + Y1@gw2 + Y2@gw3 + gb
    gemm(h_ablk, gw1bsw + (size_t)l * DDIM * DDIM / 2, zf, nullptr, gb + (size_t)l * DDIM, NP, DDIM, 0, 1);
    gemm(Y1ablk, gw2bsw + (size_t)l * DDIM * DDIM / 2, zf, nullptr, nullptr, NP, DDIM, 1, 1);
    gemm(Y2ablk, gw3bsw + (size_t)l * DDIM * DDIM / 2, zf, nullptr, nullptr, NP, DDIM, 1, 1);

    // attention
    node_pq<<<NNODE / 8, 256, 0, stream>>>(zf, attn + (size_t)l * 257, pbuf, qbuf);
    zero_f32<<<NDB, 256, 0, stream>>>(hout, ND);
    zero_f32<<<(NNODE + 255) / 256, 256, 0, stream>>>(denom, NNODE);
    fill_u32<<<(NNODE + 255) / 256, 256, 0, stream>>>(menc, 0x007FFFFFu, NNODE); // enc(-inf)
    edge_score<<<EB, 256, 0, stream>>>(src, dst, e_arr, wprod, l, pbuf, qbuf,
                                       attn + (size_t)l * 257, score, menc);
    edge_exp<<<EB, 256, 0, stream>>>(dst, score, menc, ewbuf, denom);
    edge_attn<<<EB, 256, 0, stream>>>(dst, ewbuf, denom, attn_e);
    edge_scatter<<<(int)(((long)EDGES * DDIM) / 256), 256, 0, stream>>>(
        src, dst, e_arr, wprod, l, attn_e, zf, wez + (size_t)l * DDIM, hout);

    // hi = relu( (0.3*g + 0.7*h0) @ ((1-beta)I + beta*wl) )
    mix_ablk<<<HDB, 256, 0, stream>>>(hout, h0f, mixb);
    gemm(mixb, Mlbsw, hf, hbsw, nullptr, NP, DDIM, 2, 1);
    cvt_ablk<<<HDB, 256, 0, stream>>>(hf, hablk);
  }

  copy_out<<<((NNODE * DDIM) + 255) / 256, 256, 0, stream>>>(hf, (float*)d_out);
}
